// LocalTransformerDecoder_51144470560901
// MI455X (gfx1250) — matrix operand order repaired and checked by reading, not compiled, not run on hardware
//
#include <hip/hip_runtime.h>
#include <hip/hip_bf16.h>
#include <stdint.h>

// ---------------- problem constants ----------------
#define BB   2
#define SS   2048
#define EE   512
#define HH   8
#define DHD  64
#define LL   6
#define MMD  2048
#define VV   32000
#define NBK  32      // number of blocks along sequence
#define BSK  64      // block size
#define NEGF (-1000000000.0f)

// ---------------- WMMA plumbing ----------------
typedef __attribute__((ext_vector_type(16))) __bf16 bf16x16;
typedef __attribute__((ext_vector_type(8)))  float  f32x8;

union Frag { uint4 q[2]; bf16x16 v; };

__device__ inline f32x8 wmma_bf16(bf16x16 a, bf16x16 b, f32x8 c) {
  // D = A(16x32 bf16) x B(32x16 bf16) + C(16x16 f32)
  return __builtin_amdgcn_wmma_f32_16x16x32_bf16(
      /*neg_a=*/false, a, /*neg_b=*/false, b,
      /*c_mod=*/(short)0, c, /*reuse_a=*/false, /*reuse_b=*/false);
}

__device__ inline uint16_t f2bf(float f) {
  union { float f; uint32_t u; } v; v.f = f;
  uint32_t u = v.u;
  return (uint16_t)((u + 0x7FFFu + ((u >> 16) & 1u)) >> 16); // RNE
}

// A fragment: row-major bf16 A, lda = K. contiguous k = 16*half + s per lane, same order as B.
__device__ inline bf16x16 load_a_frag(const uint16_t* A, int lda, int row0, int k0, int lane) {
  int half = lane >> 4;
  int row  = row0 + (lane & 15);
  const uint16_t* p = A + (size_t)row * lda + k0 + half * 16;
  Frag f;
  f.q[0] = *(const uint4*)p;         // K = k0 + half*16 + 0..7
  f.q[1] = *(const uint4*)(p + 8);   // K = k0 + half*16 + 8..15
  return f.v;
}

// B fragment: Bt is [N,K] row-major bf16 (pre-transposed weight), ldb = K.
__device__ inline bf16x16 load_b_frag(const uint16_t* Bt, int ldb, int n0, int k0, int lane) {
  int half = lane >> 4;
  int n = n0 + (lane & 15);
  const uint16_t* p = Bt + (size_t)n * ldb + k0 + half * 16;
  Frag f;
  f.q[0] = *(const uint4*)p;        // K = k0 + half*16 + 0..7
  f.q[1] = *(const uint4*)(p + 8);  // K = k0 + half*16 + 8..15
  return f.v;
}

// ---------------- weight pack: f32 [K,N] -> bf16 [N,K] ----------------
__global__ __launch_bounds__(256)
void pack_wT_kernel(const float* __restrict__ W, uint16_t* __restrict__ Bt, int K, int N) {
  int idx = blockIdx.x * 256 + threadIdx.x;
  if (idx >= K * N) return;
  int k = idx / N, n = idx - k * N;
  Bt[(size_t)n * K + k] = f2bf(W[idx]);
}

// ---------------- embedding + posemb + pad ----------------
__global__ __launch_bounds__(256)
void embed_kernel(const int* __restrict__ tokens, const float* __restrict__ emb,
                  float* __restrict__ h, float* __restrict__ pad) {
  int idx = blockIdx.x * 256 + threadIdx.x;
  if (idx >= BB * SS * EE) return;
  int e  = idx & (EE - 1);
  int bs = idx >> 9;             // EE = 512
  int s  = bs & (SS - 1);
  int b  = bs >> 11;             // SS = 2048
  int tok = (s == 0) ? 0 : tokens[b * SS + s - 1];  // shift-right with BOS=0
  int i2 = e & ~1;
  float div = __expf(-(float)i2 * (logf(10000.0f) / (float)EE));
  float ang = (float)s * div;
  float pe  = (e & 1) ? __cosf(ang) : __sinf(ang);
  h[idx] = emb[(size_t)tok * EE + e] + pe;
  if (e == 0) pad[bs] = (tokens[bs] > 0) ? 1.0f : 0.0f;
}

// ---------------- LayerNorm (row of 512) -> bf16 ----------------
__global__ __launch_bounds__(256)
void layernorm_kernel(const float* __restrict__ x, const float* __restrict__ sc,
                      const float* __restrict__ bi, uint16_t* __restrict__ y) {
  __shared__ float red[256];
  int row = blockIdx.x;
  int t = threadIdx.x;
  const float* xr = x + (size_t)row * EE;
  float v0 = xr[t], v1 = xr[t + 256];
  red[t] = v0 + v1; __syncthreads();
  for (int o = 128; o > 0; o >>= 1) { if (t < o) red[t] += red[t + o]; __syncthreads(); }
  float mu = red[0] * (1.0f / EE);
  __syncthreads();
  float d0 = v0 - mu, d1 = v1 - mu;
  red[t] = d0 * d0 + d1 * d1; __syncthreads();
  for (int o = 128; o > 0; o >>= 1) { if (t < o) red[t] += red[t + o]; __syncthreads(); }
  float rstd = rsqrtf(red[0] * (1.0f / EE) + 1e-6f);
  uint16_t* yr = y + (size_t)row * EE;
  yr[t]       = f2bf(d0 * rstd * sc[t]       + bi[t]);
  yr[t + 256] = f2bf(d1 * rstd * sc[t + 256] + bi[t + 256]);
}

// ---------------- generic WMMA GEMM: C[M,N] = act((alpha*A*Bt^T + bias)*rowScale) (+res) ----
__global__ __launch_bounds__(256)
void gemm_wmma_kernel(const uint16_t* __restrict__ A, const uint16_t* __restrict__ Bt,
                      float* __restrict__ Cf, uint16_t* __restrict__ Cbf,
                      const float* __restrict__ bias, const float* __restrict__ rowScale,
                      const float* __restrict__ residual,
                      float alpha, int act, int M, int N, int K) {
  int lane = threadIdx.x & 31;
  int wave = threadIdx.x >> 5;
  int wm = wave >> 2, wn = wave & 3;           // 2x4 waves -> 32x256 block tile
  int row0 = blockIdx.y * 32 + wm * 16;
  int col0 = blockIdx.x * 256 + wn * 64;       // wave tile: 16 x 64
  if (row0 >= M || col0 >= N) return;

  f32x8 acc[4];
#pragma unroll
  for (int j = 0; j < 4; ++j)
#pragma unroll
    for (int v = 0; v < 8; ++v) acc[j][v] = 0.0f;

  for (int k0 = 0; k0 < K; k0 += 32) {
    bf16x16 a = load_a_frag(A, K, row0, k0, lane);
#pragma unroll
    for (int j = 0; j < 4; ++j) {
      bf16x16 b = load_b_frag(Bt, K, col0 + j * 16, k0, lane);
      acc[j] = wmma_bf16(a, b, acc[j]);
    }
  }

  int half = lane >> 4;
  int cRow0 = row0 + half * 8;
#pragma unroll
  for (int j = 0; j < 4; ++j) {
    int col = col0 + j * 16 + (lane & 15);
    float bcol = bias ? bias[col] : 0.0f;
#pragma unroll
    for (int v = 0; v < 8; ++v) {
      int row = cRow0 + v;
      float x = acc[j][v] * alpha + bcol;
      if (rowScale) x *= rowScale[row];
      if (act == 1) { // tanh-approx GELU (jax.nn.gelu default)
        float x3 = x * x * x;
        x = 0.5f * x * (1.0f + tanhf(0.7978845608028654f * (x + 0.044715f * x3)));
      }
      size_t o = (size_t)row * N + col;
      if (residual) x += residual[o];
      if (Cf)  Cf[o]  = x;
      if (Cbf) Cbf[o] = f2bf(x);
    }
  }
}

// ---------------- ksum: mean of k over each block ----------------
__global__ __launch_bounds__(256)
void ksum_kernel(const float* __restrict__ k, float* __restrict__ ksum) {
  int idx = blockIdx.x * 256 + threadIdx.x;      // [b][n][h][d]
  if (idx >= BB * NBK * HH * DHD) return;
  int d = idx & 63, h = (idx >> 6) & 7, n = (idx >> 9) & 31, b = idx >> 14;
  const float* base = k + (size_t)(b * SS + n * BSK) * EE + h * DHD + d;
  float acc = 0.0f;
  for (int j = 0; j < BSK; ++j) acc += base[(size_t)j * EE];
  ksum[idx] = acc * (1.0f / BSK);
}

// ---------------- sinkhorn: one WG per (b,h); 32x32 logits in LDS ----------------
__global__ __launch_bounds__(256)
void sinkhorn_kernel(const float* __restrict__ ksum, const float* __restrict__ sortw,
                     const float* __restrict__ sortb, float* __restrict__ R) {
  __shared__ float lg[NBK][NBK];
  int bh = blockIdx.x;              // b*HH + h
  int b = bh >> 3, h = bh & 7;
  int t = threadIdx.x;
#pragma unroll
  for (int i = 0; i < 4; ++i) {
    int idx = t + 256 * i;
    int n = idx >> 5, m = idx & 31;
    float acc = sortb[h * NBK + m];
    const float* ks = ksum + (size_t)((b * NBK + n) * HH + h) * DHD;
    const float* sw = sortw + (size_t)h * DHD * NBK + m;
    for (int d = 0; d < DHD; ++d) acc += ks[d] * sw[(size_t)d * NBK];
    lg[n][m] = (n > m) ? acc : NEGF;
  }
  __syncthreads();
  for (int it = 0; it < 5; ++it) {
    if (t < NBK) {                  // row logsumexp over m
      float mx = -1e30f;
      for (int m = 0; m < NBK; ++m) mx = fmaxf(mx, lg[t][m]);
      float sm = 0.0f;
      for (int m = 0; m < NBK; ++m) sm += __expf(lg[t][m] - mx);
      float lse = mx + __logf(sm);
      for (int m = 0; m < NBK; ++m) lg[t][m] -= lse;
    }
    __syncthreads();
    if (t < NBK) {                  // col logsumexp over n
      float mx = -1e30f;
      for (int n = 0; n < NBK; ++n) mx = fmaxf(mx, lg[n][t]);
      float sm = 0.0f;
      for (int n = 0; n < NBK; ++n) sm += __expf(lg[n][t] - mx);
      float lse = mx + __logf(sm);
      for (int n = 0; n < NBK; ++n) lg[n][t] -= lse;
    }
    __syncthreads();
  }
#pragma unroll
  for (int i = 0; i < 4; ++i) {
    int idx = t + 256 * i;
    int n = idx >> 5, m = idx & 31;
    R[(size_t)bh * (NBK * NBK) + idx] = (n > m) ? __expf(lg[n][m]) : 0.0f;
  }
}

// ---------------- stage: build bf16 tiles qb/kb/k_srt (row-major) and vT/v_srtT ----
__global__ __launch_bounds__(256)
void stage_kernel(const float* __restrict__ q, const float* __restrict__ k,
                  const float* __restrict__ v, const float* __restrict__ R,
                  uint16_t* __restrict__ qb, uint16_t* __restrict__ kb,
                  uint16_t* __restrict__ ksrt, uint16_t* __restrict__ vT,
                  uint16_t* __restrict__ vsT) {
  __shared__ float r[NBK];
  int tile = blockIdx.x;            // (b*NBK + n)*HH + h
  int h = tile & 7, n = (tile >> 3) & 31, b = tile >> 8;
  int t = threadIdx.x;
  if (t < NBK) r[t] = R[(size_t)((b * HH + h) * NBK + n) * NBK + t];
  __syncthreads();
  size_t tb = (size_t)tile * 4096;
  for (int i = 0; i < 16; ++i) {
    int idx = t + 256 * i;
    int j = idx >> 6, d = idx & 63;
    size_t g = (size_t)(b * SS + n * BSK + j) * EE + h * DHD + d;
    qb[tb + idx] = f2bf(q[g]);
    kb[tb + idx] = f2bf(k[g]);
    vT[tb + d * 64 + j] = f2bf(v[g]);
    float ks = 0.0f, vs = 0.0f;
    for (int m = 0; m < n; ++m) {   // R is strictly lower-triangular
      size_t gm = (size_t)(b * SS + m * BSK + j) * EE + h * DHD + d;
      ks += r[m] * k[gm];
      vs += r[m] * v[gm];
    }
    ksrt[tb + idx] = f2bf(ks);
    vsT[tb + d * 64 + j] = f2bf(vs);
  }
}

// ---------------- attention core: WG per (b,n,h); WMMA QK^T + softmax + WMMA PV ----
__global__ __launch_bounds__(256)
void attn_kernel(const uint16_t* __restrict__ qb, const uint16_t* __restrict__ kb,
                 const uint16_t* __restrict__ ksrt, const uint16_t* __restrict__ vT,
                 const uint16_t* __restrict__ vsT, const float* __restrict__ pad,
                 uint16_t* __restrict__ obf) {
  __shared__ __align__(16) uint16_t sQ[4096], sK[4096], sKs[4096], sVT[4096], sVsT[4096];
  __shared__ float sS[64 * 128];
  __shared__ __align__(16) uint16_t sP[64 * 128];

  int tile = blockIdx.x;
  int h = tile & 7, n = (tile >> 3) & 31, b = tile >> 8;
  int t = threadIdx.x;
  size_t tb = (size_t)tile * 4096;

  { // stage tiles into LDS (8B vector copies)
    const uint2* g0 = (const uint2*)(qb + tb);  uint2* s0 = (uint2*)sQ;
    const uint2* g1 = (const uint2*)(kb + tb);  uint2* s1 = (uint2*)sK;
    const uint2* g2 = (const uint2*)(ksrt + tb);uint2* s2 = (uint2*)sKs;
    const uint2* g3 = (const uint2*)(vT + tb);  uint2* s3 = (uint2*)sVT;
    const uint2* g4 = (const uint2*)(vsT + tb); uint2* s4 = (uint2*)sVsT;
    for (int i = t; i < 1024; i += 256) {
      s0[i] = g0[i]; s1[i] = g1[i]; s2[i] = g2[i]; s3[i] = g3[i]; s4[i] = g4[i];
    }
  }
  __syncthreads();

  int lane = t & 31, wave = t >> 5;
  int half = lane >> 4, l15 = lane & 15;

  // ---- scores: 64x128 = [q.kb | q.k_srt], 32 tiles over 8 waves ----
  for (int tt = wave; tt < 32; tt += 8) {
    int mt = tt >> 3, nt = tt & 7;
    f32x8 acc;
#pragma unroll
    for (int v = 0; v < 8; ++v) acc[v] = 0.0f;
    const uint16_t* Bsrc = (nt < 4) ? sK : sKs;
    int ncol = (nt & 3) * 16 + l15;
    int arow = mt * 16 + l15;
#pragma unroll
    for (int ks = 0; ks < 2; ++ks) {
      int k0 = ks * 32;
      Frag fa, fb;
      const uint16_t* pa = sQ + arow * 64 + k0 + half * 16;
      fa.q[0] = *(const uint4*)pa; fa.q[1] = *(const uint4*)(pa + 8);
      const uint16_t* pb = Bsrc + ncol * 64 + k0 + half * 16;
      fb.q[0] = *(const uint4*)pb; fb.q[1] = *(const uint4*)(pb + 8);
      acc = wmma_bf16(fa.v, fb.v, acc);
    }
    int col = nt * 16 + l15;     // 0..127
#pragma unroll
    for (int v = 0; v < 8; ++v) {
      int row = mt * 16 + half * 8 + v;
      float x = acc[v];
      if (col < 64) {            // local: causal + key pad mask
        bool ok = (row >= col) && (pad[(size_t)b * SS + n * BSK + col] > 0.0f);
        x = ok ? x : NEGF;
      } else {                   // sorted stream masked for first block
        x = (n > 0) ? x : NEGF;
      }
      sS[row * 128 + col] = x;
    }
  }
  __syncthreads();

  // ---- softmax over 128, 4 lanes per row ----
  {
    int row = t >> 2, part = t & 3;
    const float* sr = sS + row * 128 + part * 32;
    float mx = -1e30f;
    for (int c = 0; c < 32; ++c) mx = fmaxf(mx, sr[c]);
    mx = fmaxf(mx, __shfl_xor(mx, 1, 32));
    mx = fmaxf(mx, __shfl_xor(mx, 2, 32));
    float sm = 0.0f;
    for (int c = 0; c < 32; ++c) sm += __expf(sr[c] - mx);
    sm += __shfl_xor(sm, 1, 32);
    sm += __shfl_xor(sm, 2, 32);
    float inv = 1.0f / sm;
    uint16_t* pr = sP + row * 128 + part * 32;
    for (int c = 0; c < 32; ++c) pr[c] = f2bf(__expf(sr[c] - mx) * inv);
  }
  __syncthreads();

  // ---- O = P[64,128] x [vb; v_srt][128,64]; 16 tiles over 8 waves ----
  for (int tt = wave; tt < 16; tt += 8) {
    int mt = tt >> 2, nt = tt & 3;
    f32x8 acc;
#pragma unroll
    for (int v = 0; v < 8; ++v) acc[v] = 0.0f;
    int arow = mt * 16 + l15;
    int ncol = nt * 16 + l15;   // output dim d
#pragma unroll
    for (int ks = 0; ks < 4; ++ks) {
      int k0 = ks * 32;
      Frag fa, fb;
      const uint16_t* pa = sP + arow * 128 + k0 + half * 16;
      fa.q[0] = *(const uint4*)pa; fa.q[1] = *(const uint4*)(pa + 8);
      int kk = k0 + half * 16;   // 16-element run stays within one V source
      const uint16_t* pb = (kk < 64) ? (sVT + ncol * 64 + kk)
                                     : (sVsT + ncol * 64 + (kk - 64));
      fb.q[0] = *(const uint4*)pb; fb.q[1] = *(const uint4*)(pb + 8);
      acc = wmma_bf16(fa.v, fb.v, acc);
    }
#pragma unroll
    for (int v = 0; v < 8; ++v) {
      int row = mt * 16 + half * 8 + v;
      size_t g = (size_t)(b * SS + n * BSK + row) * EE + h * DHD + ncol;
      obf[g] = f2bf(acc[v]);
    }
  }
}

// ---------------- host-side orchestration ----------------
extern "C" void kernel_launch(void* const* d_in, const int* in_sizes, int n_in,
                              void* d_out, int out_size, void* d_ws, size_t ws_size,
                              hipStream_t stream) {
  const int*   tokens = (const int*)  d_in[0];
  const float* embedw = (const float*)d_in[1];
  const float* ln1_s  = (const float*)d_in[2];
  const float* ln1_b  = (const float*)d_in[3];
  const float* wq     = (const float*)d_in[4];
  const float* wk     = (const float*)d_in[5];
  const float* wv     = (const float*)d_in[6];
  const float* wo     = (const float*)d_in[7];
  const float* sortw  = (const float*)d_in[8];
  const float* sortb  = (const float*)d_in[9];
  const float* ln2_s  = (const float*)d_in[10];
  const float* ln2_b  = (const float*)d_in[11];
  const float* mlp_w1 = (const float*)d_in[12];
  const float* mlp_b1 = (const float*)d_in[13];
  const float* mlp_w2 = (const float*)d_in[14];
  const float* mlp_b2 = (const float*)d_in[15];
  const float* lnf_s  = (const float*)d_in[16];
  const float* lnf_b  = (const float*)d_in[17];
  const float* out_w  = (const float*)d_in[18];
  const float* out_b  = (const float*)d_in[19];
  float* out = (float*)d_out;

  uint8_t* ws = (uint8_t*)d_ws;
  size_t off = 0;
  auto alloc = [&](size_t bytes) -> void* {
    void* p = ws + off;
    off += (bytes + 255) & ~(size_t)255;
    return p;
  };
  const int MROWS = BB * SS;                       // 4096
  float*    hbuf  = (float*)   alloc((size_t)MROWS * EE * 4);
  uint16_t* xln   = (uint16_t*)alloc((size_t)MROWS * EE * 2);
  float*    qf    = (float*)   alloc((size_t)MROWS * EE * 4);
  float*    kf    = (float*)   alloc((size_t)MROWS * EE * 4);
  float*    vf    = (float*)   alloc((size_t)MROWS * EE * 4);
  float*    pad   = (float*)   alloc((size_t)MROWS * 4);
  float*    ksum  = (float*)   alloc((size_t)BB * NBK * HH * DHD * 4);
  float*    Rbuf  = (float*)   alloc((size_t)BB * HH * NBK * NBK * 4);
  uint16_t* wT    = (uint16_t*)alloc((size_t)EE * MMD * 2);         // largest layer weight
  uint16_t* outwT = (uint16_t*)alloc((size_t)VV * EE * 2);
  uint16_t* mid   = (uint16_t*)alloc((size_t)MROWS * MMD * 2);
  uint16_t* obf   = (uint16_t*)alloc((size_t)MROWS * EE * 2);
  uint16_t* qb    = (uint16_t*)alloc((size_t)MROWS * EE * 2);
  uint16_t* kb    = (uint16_t*)alloc((size_t)MROWS * EE * 2);
  uint16_t* ksb   = (uint16_t*)alloc((size_t)MROWS * EE * 2);
  uint16_t* vTb   = (uint16_t*)alloc((size_t)MROWS * EE * 2);
  uint16_t* vsTb  = (uint16_t*)alloc((size_t)MROWS * EE * 2);

  auto pack = [&](const float* W, uint16_t* Bt, int K, int N) {
    int total = K * N;
    pack_wT_kernel<<<(total + 255) / 256, 256, 0, stream>>>(W, Bt, K, N);
  };
  auto gemm = [&](const uint16_t* A, const uint16_t* Bt, float* Cf, uint16_t* Cbf,
                  const float* bias, const float* rowScale, const float* residual,
                  float alpha, int act, int M, int N, int K) {
    dim3 grid(N / 256, M / 32);
    gemm_wmma_kernel<<<grid, 256, 0, stream>>>(A, Bt, Cf, Cbf, bias, rowScale,
                                               residual, alpha, act, M, N, K);
  };

  embed_kernel<<<(BB * SS * EE + 255) / 256, 256, 0, stream>>>(tokens, embedw, hbuf, pad);
  pack(out_w, outwT, EE, VV);   // pack vocab projection once per call

  for (int l = 0; l < LL; ++l) {
    layernorm_kernel<<<MROWS, 256, 0, stream>>>(hbuf, ln1_s + l * EE, ln1_b + l * EE, xln);

    pack(wq + (size_t)l * EE * EE, wT, EE, EE);
    gemm(xln, wT, qf, nullptr, nullptr, nullptr, nullptr, 0.125f, 0, MROWS, EE, EE);
    pack(wk + (size_t)l * EE * EE, wT, EE, EE);
    gemm(xln, wT, kf, nullptr, nullptr, pad, nullptr, 1.0f, 0, MROWS, EE, EE);
    pack(wv + (size_t)l * EE * EE, wT, EE, EE);
    gemm(xln, wT, vf, nullptr, nullptr, pad, nullptr, 1.0f, 0, MROWS, EE, EE);

    ksum_kernel<<<(BB * NBK * HH * DHD + 255) / 256, 256, 0, stream>>>(kf, ksum);
    sinkhorn_kernel<<<BB * HH, 256, 0, stream>>>(
        ksum, sortw + (size_t)l * HH * DHD * NBK, sortb + (size_t)l * HH * NBK, Rbuf);
    stage_kernel<<<BB * NBK * HH, 256, 0, stream>>>(qf, kf, vf, Rbuf, qb, kb, ksb, vTb, vsTb);
    attn_kernel<<<BB * NBK * HH, 256, 0, stream>>>(qb, kb, ksb, vTb, vsTb, pad, obf);

    pack(wo + (size_t)l * EE * EE, wT, EE, EE);
    gemm(obf, wT, hbuf, nullptr, nullptr, nullptr, hbuf, 1.0f, 0, MROWS, EE, EE);

    layernorm_kernel<<<MROWS, 256, 0, stream>>>(hbuf, ln2_s + l * EE, ln2_b + l * EE, xln);
    pack(mlp_w1 + (size_t)l * EE * MMD, wT, EE, MMD);
    gemm(xln, wT, nullptr, mid, mlp_b1 + (size_t)l * MMD, nullptr, nullptr, 1.0f, 1,
         MROWS, MMD, EE);
    pack(mlp_w2 + (size_t)l * MMD * EE, wT, MMD, EE);
    gemm(mid, wT, hbuf, nullptr, mlp_b2 + (size_t)l * EE, nullptr, hbuf, 1.0f, 0,
         MROWS, EE, MMD);
  }

  layernorm_kernel<<<MROWS, 256, 0, stream>>>(hbuf, lnf_s, lnf_b, xln);
  gemm(xln, outwT, out, nullptr, out_b, nullptr, nullptr, 1.0f, 0, MROWS, VV, EE);
}
